// ScaledDotProductAttention_84567906058715
// MI455X (gfx1250) — matrix operand order repaired and checked by reading, not compiled, not run on hardware
//
#include <hip/hip_runtime.h>
#include <hip/hip_bf16.h>

// CDNA5 (gfx1250, wave32) flash attention, bf16 WMMA with f32 accumulation.
// B=2, H=16, S=2048, D=64 (from the reference setup_inputs()).

typedef __attribute__((ext_vector_type(16))) __bf16 v16bf;
typedef __attribute__((ext_vector_type(8)))  float  v8f;

#define SDIM   2048
#define DDIM   64
#define QTILE  64            // query rows per block (16 per wave)
#define KTILE  64            // keys per iteration
#define NWAVES 4
#define LDS_PAD 4
#define LROW  (DDIM + LDS_PAD)   // padded row length (bf16 elements), 68 -> 136B stride

// Load 16 contiguous bf16 (8-byte aligned) from LDS into a B/A fragment vector.
__device__ __forceinline__ v16bf ld16bf_contig(const __bf16* p) {
    union { v16bf v; uint2 q[4]; } u;
    const uint2* s = (const uint2*)p;
    u.q[0] = s[0]; u.q[1] = s[1]; u.q[2] = s[2]; u.q[3] = s[3];
    return u.v;
}

// A-matrix fragment: two contiguous runs of 8 bf16 (K = c*32+h*8 .. and c*32+16+h*8 ..)
__device__ __forceinline__ v16bf ld_a_frag(const __bf16* run0, const __bf16* run1) {
    union { v16bf v; uint2 q[4]; } u;
    u.q[0] = *(const uint2*)(run0);
    u.q[1] = *(const uint2*)(run0 + 4);
    u.q[2] = *(const uint2*)(run1);
    u.q[3] = *(const uint2*)(run1 + 4);
    return u.v;
}

// ---- intra-16-lane xor butterflies on the VALU (v_permlane16_b32) ----
#if __has_builtin(__builtin_amdgcn_permlane16)
__device__ __forceinline__ float lane_xor16(float x, unsigned s0, unsigned s1) {
    unsigned u = __builtin_bit_cast(unsigned, x);
    unsigned r = __builtin_amdgcn_permlane16(u, u, s0, s1, false, false);
    return __builtin_bit_cast(float, r);
}
// nibble tables: lane i (within its 16-row) sources lane i^m
#define XOR1_S0 0x67452301u
#define XOR1_S1 0xEFCDAB89u
#define XOR2_S0 0x54761032u
#define XOR2_S1 0xDCFE98BAu
#define XOR4_S0 0x32107654u
#define XOR4_S1 0xBA98FEDCu
#define XOR8_S0 0xFEDCBA98u
#define XOR8_S1 0x76543210u
__device__ __forceinline__ float redmax16(float x) {
    x = fmaxf(x, lane_xor16(x, XOR1_S0, XOR1_S1));
    x = fmaxf(x, lane_xor16(x, XOR2_S0, XOR2_S1));
    x = fmaxf(x, lane_xor16(x, XOR4_S0, XOR4_S1));
    x = fmaxf(x, lane_xor16(x, XOR8_S0, XOR8_S1));
    return x;
}
__device__ __forceinline__ float redsum16(float x) {
    x += lane_xor16(x, XOR1_S0, XOR1_S1);
    x += lane_xor16(x, XOR2_S0, XOR2_S1);
    x += lane_xor16(x, XOR4_S0, XOR4_S1);
    x += lane_xor16(x, XOR8_S0, XOR8_S1);
    return x;
}
#else
__device__ __forceinline__ float redmax16(float x) {
    x = fmaxf(x, __shfl_xor(x, 1));
    x = fmaxf(x, __shfl_xor(x, 2));
    x = fmaxf(x, __shfl_xor(x, 4));
    x = fmaxf(x, __shfl_xor(x, 8));
    return x;
}
__device__ __forceinline__ float redsum16(float x) {
    x += __shfl_xor(x, 1);
    x += __shfl_xor(x, 2);
    x += __shfl_xor(x, 4);
    x += __shfl_xor(x, 8);
    return x;
}
#endif

// base-2 exponential -> v_exp_f32
__device__ __forceinline__ float fast_exp2(float x) {
#if __has_builtin(__builtin_amdgcn_exp2f)
    return __builtin_amdgcn_exp2f(x);
#else
    return exp2f(x);
#endif
}

__global__ __launch_bounds__(128)
void fa_fwd_gfx1250(const float* __restrict__ Q, const float* __restrict__ K,
                    const float* __restrict__ V, float* __restrict__ O)
{
    __shared__ __bf16 KS[KTILE][LROW];          // K tile, [key][d]   (for Q*K^T B-frags)
    __shared__ __bf16 VT[DDIM][LROW];           // V tile transposed, [d][key] (for P*V B-frags)
    __shared__ __bf16 PS[NWAVES][16][LROW];     // per-wave P scratch, [m][key]

    const int tid  = threadIdx.x;
    const int wave = tid >> 5;
    const int lane = tid & 31;
    const int lm   = lane & 15;   // column index / row-within-tile
    const int h    = lane >> 4;   // half-wave select (ISA fragment layouts)

    const int bh = blockIdx.y;                          // 0..B*H-1
    const int q0 = blockIdx.x * QTILE + wave * 16;      // first query row of this wave
    const size_t base = (size_t)bh * SDIM * DDIM;

    // softmax in exp2 domain: fold 1/sqrt(64) * log2(e) into Q
    const float scale = 0.125f * 1.44269504088896340736f;

    // ---- Q A-fragments (16x32 bf16 per chunk), scale folded in (fp32 first) ----
    v16bf aQ[2];
    {
        const float* qrow = Q + base + (size_t)(q0 + lm) * DDIM;
        #pragma unroll
        for (int c = 0; c < 2; ++c) {
            const float* r0 = qrow + c * 32 + h * 8;
            const float* r1 = qrow + c * 32 + 16 + h * 8;
            float4 a0 = *(const float4*)(r0);
            float4 a1 = *(const float4*)(r0 + 4);
            float4 b0 = *(const float4*)(r1);
            float4 b1 = *(const float4*)(r1 + 4);
            aQ[c][0]  = (__bf16)(a0.x * scale); aQ[c][1]  = (__bf16)(a0.y * scale);
            aQ[c][2]  = (__bf16)(a0.z * scale); aQ[c][3]  = (__bf16)(a0.w * scale);
            aQ[c][4]  = (__bf16)(a1.x * scale); aQ[c][5]  = (__bf16)(a1.y * scale);
            aQ[c][6]  = (__bf16)(a1.z * scale); aQ[c][7]  = (__bf16)(a1.w * scale);
            aQ[c][8]  = (__bf16)(b0.x * scale); aQ[c][9]  = (__bf16)(b0.y * scale);
            aQ[c][10] = (__bf16)(b0.z * scale); aQ[c][11] = (__bf16)(b0.w * scale);
            aQ[c][12] = (__bf16)(b1.x * scale); aQ[c][13] = (__bf16)(b1.y * scale);
            aQ[c][14] = (__bf16)(b1.z * scale); aQ[c][15] = (__bf16)(b1.w * scale);
        }
    }

    // ---- flash state ----
    v8f acc[4];
    #pragma unroll
    for (int n = 0; n < 4; ++n) acc[n] = (v8f){};
    float mrow[8], lrow[8];
    #pragma unroll
    for (int r = 0; r < 8; ++r) { mrow[r] = -INFINITY; lrow[r] = 0.0f; }

    const int NT = SDIM / KTILE;
    for (int kt = 0; kt < NT; ++kt) {
        __syncthreads();  // previous iteration's LDS reads done

        // ---- cooperative K/V tile load: f32 -> bf16, V transposed ----
        const float* Kg = K + base + (size_t)kt * KTILE * DDIM;
        const float* Vg = V + base + (size_t)kt * KTILE * DDIM;
        #pragma unroll
        for (int e = tid * 4; e < KTILE * DDIM; e += 128 * 4) {
            const int key = e >> 6;     // /DDIM
            const int d   = e & 63;
            float4 k4 = *(const float4*)(Kg + e);
            float4 v4 = *(const float4*)(Vg + e);
            // packed 4 x bf16 -> single 8-byte LDS store
            union { uint2 u; __bf16 b[4]; } pk;
            pk.b[0] = (__bf16)k4.x; pk.b[1] = (__bf16)k4.y;
            pk.b[2] = (__bf16)k4.z; pk.b[3] = (__bf16)k4.w;
            *(uint2*)&KS[key][d] = pk.u;
            VT[d + 0][key] = (__bf16)v4.x;
            VT[d + 1][key] = (__bf16)v4.y;
            VT[d + 2][key] = (__bf16)v4.z;
            VT[d + 3][key] = (__bf16)v4.w;
            if (kt + 1 < NT) {  // overlap next tile's HBM fetch with this tile's compute
                __builtin_prefetch(Kg + KTILE * DDIM + e, 0, 1);
                __builtin_prefetch(Vg + KTILE * DDIM + e, 0, 1);
            }
        }
        __syncthreads();

        // ---- scores: S(16x64) = Qhat(16x64) * K^T ; 4 tiles x 2 K-chunks ----
        v8f s[4];
        #pragma unroll
        for (int n = 0; n < 4; ++n) {
            v8f cacc = (v8f){};
            #pragma unroll
            for (int c = 0; c < 2; ++c) {
                // B-frag (32x16): lane = key column n*16+lm, K runs d = c*32+h*8 and c*32+16+h*8
                v16bf bK = ld_a_frag(&KS[n * 16 + lm][c * 32 + h * 8], &KS[n * 16 + lm][c * 32 + 16 + h * 8]);
                cacc = __builtin_amdgcn_wmma_f32_16x16x32_bf16(
                           false, aQ[c], false, bK, (short)0, cacc, false, false);
            }
            s[n] = cacc;
        }

        // ---- online softmax update (exp2 domain) ----
        float mnew[8], alpha[8];
        #pragma unroll
        for (int r = 0; r < 8; ++r) {
            float mx = fmaxf(fmaxf(s[0][r], s[1][r]), fmaxf(s[2][r], s[3][r]));
            mx = redmax16(mx);
            mnew[r]  = fmaxf(mrow[r], mx);
            alpha[r] = fast_exp2(mrow[r] - mnew[r]);
            mrow[r]  = mnew[r];
        }
        // exponentiate + stash P (bf16) into per-wave LDS scratch in row-major order
        #pragma unroll
        for (int n = 0; n < 4; ++n) {
            #pragma unroll
            for (int r = 0; r < 8; ++r) {
                float p = fast_exp2(s[n][r] - mnew[r]);
                s[n][r] = p;
                PS[wave][r + 8 * h][n * 16 + lm] = (__bf16)p;
            }
        }
        #pragma unroll
        for (int r = 0; r < 8; ++r) {
            float sm = (s[0][r] + s[1][r]) + (s[2][r] + s[3][r]);
            sm = redsum16(sm);
            lrow[r] = lrow[r] * alpha[r] + sm;
            #pragma unroll
            for (int n = 0; n < 4; ++n) acc[n][r] *= alpha[r];
        }
        __syncthreads();  // make P visible for re-layout read

        // ---- O += P(16x64) * V(64x64) ; 2 key-chunks x 4 d-tiles ----
        #pragma unroll
        for (int c = 0; c < 2; ++c) {
            // A-frag from P scratch: row lm, K runs c*32+h*8 and c*32+16+h*8
            v16bf aP = ld_a_frag(&PS[wave][lm][c * 32 + h * 8],
                                 &PS[wave][lm][c * 32 + 16 + h * 8]);
            #pragma unroll
            for (int n = 0; n < 4; ++n) {
                // B-frag (32x16): lane = d column n*16+lm, K runs key = c*32+h*8 and c*32+16+h*8 in VT
                v16bf bV = ld_a_frag(&VT[n * 16 + lm][c * 32 + h * 8], &VT[n * 16 + lm][c * 32 + 16 + h * 8]);
                acc[n] = __builtin_amdgcn_wmma_f32_16x16x32_bf16(
                             false, aP, false, bV, (short)0, acc[n], false, false);
            }
        }
    }

    // ---- normalize and write out (f32) ----
    #pragma unroll
    for (int r = 0; r < 8; ++r) {
        const float inv = 1.0f / lrow[r];
        float* orow = O + base + (size_t)(q0 + r + 8 * h) * DDIM;
        #pragma unroll
        for (int n = 0; n < 4; ++n)
            orow[n * 16 + lm] = acc[n][r] * inv;
    }
}

extern "C" void kernel_launch(void* const* d_in, const int* in_sizes, int n_in,
                              void* d_out, int out_size, void* d_ws, size_t ws_size,
                              hipStream_t stream) {
    (void)in_sizes; (void)n_in; (void)out_size; (void)d_ws; (void)ws_size;
    const float* Q = (const float*)d_in[0];
    const float* K = (const float*)d_in[1];
    const float* V = (const float*)d_in[2];
    float* O = (float*)d_out;

    const int BH = 2 * 16;
    dim3 grid(SDIM / QTILE, BH);
    dim3 block(128);
    fa_fwd_gfx1250<<<grid, block, 0, stream>>>(Q, K, V, O);
}
